// RBMSymm_Scan_22874995819121
// MI455X (gfx1250) — matrix operand order repaired and checked by reading, not compiled, not run on hardware
//
#include <hip/hip_runtime.h>
#include <stdint.h>

typedef __attribute__((ext_vector_type(16))) __bf16 v16bf;
typedef __attribute__((ext_vector_type(8)))  __bf16 v8bf;
typedef __attribute__((ext_vector_type(8)))  float  v8f;

typedef __attribute__((ext_vector_type(4))) unsigned int uint32x4;
typedef __attribute__((ext_vector_type(8))) int          int32x8;
typedef __attribute__((ext_vector_type(4))) int          int32x4;

#if defined(__has_builtin)
#if __has_builtin(__builtin_amdgcn_tensor_load_to_lds)
#define HAVE_TDM 1
#endif
#endif

#define LOG2F 0.69314718055994530942f

__device__ __forceinline__ unsigned short f2bf(float x) {
    // round-to-nearest-even f32 -> bf16
    unsigned int u = __float_as_uint(x);
    unsigned int r = u + 0x7FFFu + ((u >> 16) & 1u);
    return (unsigned short)(r >> 16);
}

__device__ __forceinline__ float logcosh_f(float x) {
    float ax = fabsf(x);
    return ax + log1pf(__expf(-2.0f * ax)) - LOG2F;
}

// ---------------------------------------------------------------------------
// Prep 1: K-major packed permuted weights.
//   WbT[(g*16+f)*N + perm[g][n]] = bf16(W[n*16+f])
// ---------------------------------------------------------------------------
__global__ void build_wbt_kernel(const float* __restrict__ W,
                                 const int* __restrict__ perms,
                                 unsigned short* __restrict__ WbT,
                                 int N, int G) {
    int idx = blockIdx.x * blockDim.x + threadIdx.x;
    if (idx >= G * N) return;
    int g = idx / N;
    int n = idx - g * N;
    int m = perms[idx];
    const float* wrow = W + (size_t)n * 16;
    size_t base = ((size_t)g * 16) * (size_t)N + (size_t)m;
    #pragma unroll
    for (int f = 0; f < 16; ++f)
        WbT[base + (size_t)f * N] = f2bf(wrow[f]);
}

// ---------------------------------------------------------------------------
// Prep 2: x f32 -> bf16 (row-major, TDM-friendly) + per-row sum(x).
// One block (256 thr) per row of 1024 floats.
// ---------------------------------------------------------------------------
__global__ __launch_bounds__(256)
void convert_x_kernel(const float* __restrict__ x,
                      unsigned short* __restrict__ xb,
                      float* __restrict__ xsum, int N) {
    __shared__ float wsum[8];
    const int row = blockIdx.x;
    const int t   = threadIdx.x;
    float4 v = ((const float4*)(x + (size_t)row * N))[t];
    unsigned int p0 = (unsigned int)f2bf(v.x) | ((unsigned int)f2bf(v.y) << 16);
    unsigned int p1 = (unsigned int)f2bf(v.z) | ((unsigned int)f2bf(v.w) << 16);
    *(uint2*)&xb[(size_t)row * N + 4 * t] = make_uint2(p0, p1);
    float s = v.x + v.y + v.z + v.w;
    #pragma unroll
    for (int m = 16; m >= 1; m >>= 1) s += __shfl_xor(s, m, 32);
    if ((t & 31) == 0) wsum[t >> 5] = s;
    __syncthreads();
    if (t == 0) {
        float tot = 0.f;
        #pragma unroll
        for (int i = 0; i < 8; ++i) tot += wsum[i];
        xsum[row] = tot;
    }
}

// ---------------------------------------------------------------------------
// Fused bf16-WMMA GEMM + bias + logcosh + column reduction + v_bias*sum(x).
// Block: 256 threads (8 wave32); block tile: 64 rows x all G*16 columns.
// A tile staged by the Tensor Data Mover with HW padding (16B gap per 512
// elements -> LDS row stride 1040 bf16, mitigates bank conflicts, keeps all
// 32-element K-blocks contiguous and 16B aligned).
// ---------------------------------------------------------------------------
__global__ __launch_bounds__(256)
void rbm_symm_fused_kernel(const unsigned short* __restrict__ xb,
                           const unsigned short* __restrict__ WbT,
                           const float* __restrict__ bvec,
                           const float* __restrict__ vbias,
                           const float* __restrict__ xsum,
                           float* __restrict__ out,
                           int G) {
    constexpr int N    = 1024;           // n_sites (K dim)
    constexpr int ROWS = 64;             // rows per block
    constexpr int ASTR = N + 16;         // LDS row stride in bf16 (2 pads x 8)

    extern __shared__ unsigned short As[];   // [ROWS][ASTR] bf16 (padded)
    __shared__ float pr[8][16];              // per-wave partial row sums

    const int t    = threadIdx.x;
    const int w    = t >> 5;
    const int lane = t & 31;
    const int row0 = blockIdx.x * ROWS;

#if defined(HAVE_TDM)
    // ---- TDM: DMA the 64x1024 bf16 tile global -> LDS with padding ----
    if (w == 0) {
        unsigned long long ga = (unsigned long long)(uintptr_t)(xb + (size_t)row0 * N);
        unsigned int lds     = (unsigned int)(uintptr_t)As;   // LDS byte offset
        uint32x4 g0;
        g0[0] = 1u;                                  // count=1 (valid user D#)
        g0[1] = lds;                                 // lds_addr
        g0[2] = (unsigned int)(ga & 0xFFFFFFFFu);    // global_addr[31:0]
        g0[3] = (unsigned int)((ga >> 32) & 0x01FFFFFFu) | 0x80000000u; // [56:32] | type=2
        int32x8 g1;
        g1[0] = (int)((1u << 16)      // data_size = 2 bytes
                    | (1u << 20)      // pad_enable
                    | (7u << 22)      // pad_interval: 256 dwords
                    | (3u << 25));    // pad_amount: 4 dwords (16B)
        g1[1] = (int)(1024u << 16);   // tensor_dim0 = 1024 (bits 79:48, low half)
        g1[2] = (int)(64u << 16);     // tensor_dim0 hi=0 | tensor_dim1 = 64
        g1[3] = (int)(1024u << 16);   // tensor_dim1 hi=0 | tile_dim0 = 1024
        g1[4] = 64;                   // tile_dim1 = 64, tile_dim2 = 0
        g1[5] = 1024;                 // tensor_dim0_stride[31:0]
        g1[6] = 0;                    // stride hi | tensor_dim1_stride lo
        g1[7] = 0;
        int32x4 gz = {0, 0, 0, 0};
#if __clang_major__ >= 23
        int32x8 gz8 = {0, 0, 0, 0, 0, 0, 0, 0};
        __builtin_amdgcn_tensor_load_to_lds(g0, g1, gz, gz, gz8, 0);
#else
        __builtin_amdgcn_tensor_load_to_lds(g0, g1, gz, gz, 0);
#endif
        __builtin_amdgcn_s_wait_tensorcnt(0);
    }
#else
    // ---- fallback: manual bf16 staging with the same padded layout ----
    for (int k = 0; k < ROWS; ++k) {
        uint2 p = ((const uint2*)(xb + (size_t)(row0 + k) * N))[t];
        *(uint2*)&As[k * ASTR + 4 * t + ((t >= 128) ? 8 : 0)] = p;
    }
#endif
    __syncthreads();

    // ---- Compute ----
    const int rt = w & 3;            // row tile within block
    const int ch = w >> 2;           // column half
    const int colTiles     = G;      // F==16 -> one 16-wide tile per group
    const int tilesPerHalf = colTiles >> 1;
    const int groups       = tilesPerHalf >> 2;   // 4 tiles per register group

    // A fragment: lanes 0-15 rows M=0..15 hold K in [0,16);
    // lanes 16-31 same rows hold K in [16,32)  (matches contiguous B order)
    const unsigned short* Ap =
        As + (rt * 16 + (lane & 15)) * ASTR + ((lane >> 4) << 4);
    const float bias = bvec[lane & 15];          // col%16 == lane%16
    const float vb   = vbias[0];

    float S[8];
    #pragma unroll
    for (int v = 0; v < 8; ++v) S[v] = 0.0f;

    for (int gi = 0; gi < groups; ++gi) {
        const int ct0 = ch * tilesPerHalf + gi * 4;
        // B fragment: lane col = ct*16 + lane%16; lanes 0-15 K=0..15,
        // lanes 16-31 K=16..31 -> 16 contiguous bf16 from K-major WbT.
        const unsigned short* B0 =
            WbT + (size_t)(ct0 * 16 + (lane & 15)) * N + ((lane >> 4) << 4);
        const unsigned short* B1 = B0 + (size_t)16 * N;
        const unsigned short* B2 = B1 + (size_t)16 * N;
        const unsigned short* B3 = B2 + (size_t)16 * N;
        __builtin_prefetch(B0 + (size_t)64 * N, 0, 1);   // next group's columns

        v8f c0 = {}, c1 = {}, c2 = {}, c3 = {};
        #pragma unroll 2
        for (int kb = 0; kb < N / 32; ++kb) {
            const unsigned short* ap = Ap + kb * 32 + ((kb & 16) ? 8 : 0);
            v8bf alo = *(const v8bf*)(ap);
            v8bf ahi = *(const v8bf*)(ap + 8);
            v16bf a = __builtin_shufflevector(alo, ahi,
                        0,1,2,3,4,5,6,7,8,9,10,11,12,13,14,15);
            v16bf b0 = *(const v16bf*)(B0 + kb * 32);
            v16bf b1 = *(const v16bf*)(B1 + kb * 32);
            v16bf b2 = *(const v16bf*)(B2 + kb * 32);
            v16bf b3 = *(const v16bf*)(B3 + kb * 32);
            c0 = __builtin_amdgcn_wmma_f32_16x16x32_bf16(false, a, false, b0, (short)0, c0, false, false);
            c1 = __builtin_amdgcn_wmma_f32_16x16x32_bf16(false, a, false, b1, (short)0, c1, false, false);
            c2 = __builtin_amdgcn_wmma_f32_16x16x32_bf16(false, a, false, b2, (short)0, c2, false, false);
            c3 = __builtin_amdgcn_wmma_f32_16x16x32_bf16(false, a, false, b3, (short)0, c3, false, false);
        }
        // C layout: VGPR v, lane l -> (M = v + 8*(l>=16), Ncol = l%16)
        #pragma unroll
        for (int v = 0; v < 8; ++v) {
            S[v] += logcosh_f(c0[v] + bias);
            S[v] += logcosh_f(c1[v] + bias);
            S[v] += logcosh_f(c2[v] + bias);
            S[v] += logcosh_f(c3[v] + bias);
        }
    }

    // Reduce over the 16 columns held in each 16-lane half.
    #pragma unroll
    for (int v = 0; v < 8; ++v) {
        float s = S[v];
        s += __shfl_xor(s, 1, 32);
        s += __shfl_xor(s, 2, 32);
        s += __shfl_xor(s, 4, 32);
        s += __shfl_xor(s, 8, 32);
        S[v] = s;
    }
    if ((lane & 15) == 0) {
        int rbase = (lane >> 4) * 8;   // lane 0 -> rows 0..7, lane 16 -> rows 8..15
        #pragma unroll
        for (int v = 0; v < 8; ++v) pr[w][rbase + v] = S[v];
    }
    __syncthreads();

    // Combine column halves + v_bias * sum(x); one writer per row -> deterministic.
    if (t < ROWS) {
        float tot = pr[t >> 4][t & 15] + pr[(t >> 4) + 4][t & 15];
        out[row0 + t] = tot + vb * xsum[row0 + t];
    }
}

extern "C" void kernel_launch(void* const* d_in, const int* in_sizes, int n_in,
                              void* d_out, int out_size, void* d_ws, size_t ws_size,
                              hipStream_t stream) {
    const float* x     = (const float*)d_in[0];
    const float* W     = (const float*)d_in[1];
    const float* b     = (const float*)d_in[2];
    const float* vbias = (const float*)d_in[3];
    const int*   perms = (const int*)d_in[4];

    const int N = 1024;                 // n_sites
    const int B = in_sizes[0] / N;      // batch
    const int G = in_sizes[4] / N;      // n_symm

    // workspace layout: [ WbT : G*16*N bf16 ][ xb : B*N bf16 ][ xsum : B f32 ]
    size_t wbtBytes = (size_t)G * 16 * N * sizeof(unsigned short);
    size_t xbBytes  = (size_t)B * N * sizeof(unsigned short);
    unsigned short* WbT = (unsigned short*)d_ws;
    unsigned short* xb  = (unsigned short*)((char*)d_ws + wbtBytes);
    float*          xsm = (float*)((char*)d_ws + wbtBytes + xbBytes);

    int prepThreads = G * N;
    build_wbt_kernel<<<(prepThreads + 255) / 256, 256, 0, stream>>>(W, perms, WbT, N, G);
    convert_x_kernel<<<B, 256, 0, stream>>>(x, xb, xsm, N);

    size_t ldsBytes = (size_t)64 * (size_t)(N + 16) * sizeof(unsigned short); // ~130 KB
    rbm_symm_fused_kernel<<<B / 64, 256, ldsBytes, stream>>>(xb, WbT, b, vbias, xsm,
                                                             (float*)d_out, G);
}
